// CapsNetLID_62002147885102
// MI455X (gfx1250) — matrix operand order repaired and checked by reading, not compiled, not run on hardware
//
#include <hip/hip_runtime.h>
#include <hip/hip_bf16.h>

// ---- vector types matching gfx1250 WMMA operand layouts ----
typedef _Float16 v16h __attribute__((ext_vector_type(16)));
typedef _Float16 v8h  __attribute__((ext_vector_type(8)));
typedef float    v8f  __attribute__((ext_vector_type(8)));

#define BATCH 64
#define LOUT 96
#define NCAPS 50
#define NROUTES 3072          // 32 * 96
#define CAPSD 8
#define OUTD 16
#define KROUTE 24576          // NROUTES * CAPSD
#define CONVK 576             // 64 * 9
#define RECON 19000           // 200 * 95

// ---------------- small utility kernels ----------------
__global__ void cvt_f16_k(const float* __restrict__ src, _Float16* __restrict__ dst, int n) {
  int i = blockIdx.x * 256 + threadIdx.x;
  if (i < n) dst[i] = (_Float16)src[i];
}

__global__ void zero_f32_k(float* __restrict__ p, int n) {
  int i = blockIdx.x * 256 + threadIdx.x;
  if (i < n) p[i] = 0.f;
}

// src (K,N) f32 row-major -> dst (N,K) f16 row-major, batched over blockIdx.z
__global__ void transpose_cvt_k(const float* __restrict__ src, _Float16* __restrict__ dst,
                                int K, int N) {
  __shared__ float tile[32][33];
  src += (size_t)blockIdx.z * K * N;
  dst += (size_t)blockIdx.z * K * N;
  int n0 = blockIdx.x * 32, k0 = blockIdx.y * 32;
  for (int j = 0; j < 32; j += 8) {
    int k = k0 + threadIdx.y + j, n = n0 + threadIdx.x;
    if (k < K && n < N) tile[threadIdx.y + j][threadIdx.x] = src[(size_t)k * N + n];
  }
  __syncthreads();
  for (int j = 0; j < 32; j += 8) {
    int n = n0 + threadIdx.y + j, k = k0 + threadIdx.x;
    if (n < N && k < K) dst[(size_t)n * K + k] = (_Float16)tile[threadIdx.x][threadIdx.y + j];
  }
}

// im2col patches: Ac[(b*96+l), ic*9+k] = emb[x[b,2l+k], ic] (f16)
__global__ void build_patches_k(const int* __restrict__ x, const float* __restrict__ emb,
                                _Float16* __restrict__ Ac) {
  int idx = blockIdx.x * 256 + threadIdx.x;
  if (idx >= BATCH * LOUT * CONVK) return;
  int row = idx / CONVK, j = idx % CONVK;
  int b = row / LOUT, l = row % LOUT;
  int ic = j / 9, kk = j % 9;
  int tok = x[b * 200 + 2 * l + kk];
  Ac[idx] = (_Float16)emb[tok * 64 + ic];
}

// ---------------- generic f16 WMMA GEMM ----------------
// C(M,N) = act( A(M,K) @ Bt(N,K)^T + bias ), f16 inputs, f32 accum.
// act: 0=none 1=relu 2=sigmoid.  umode: scatter result into u(B,3072,8) f16 layout.
// N-edge: B row pointer is CLAMPED to a valid row (garbage values, masked at store)
// so the inner loop carries no exec-mask manipulation around the WMMA.
__global__ __launch_bounds__(128) void wmma_gemm_k(
    const _Float16* __restrict__ A, const _Float16* __restrict__ Bt,
    const float* __restrict__ bias, float* __restrict__ Cf, _Float16* __restrict__ Ch,
    int M, int N, int K, int act, int umode) {
  int lane = threadIdx.x & 31;
  int wv = threadIdx.x >> 5;
  int tilesN = (N + 15) >> 4;
  int tilesM = M >> 4;
  int t = blockIdx.x * 4 + wv;
  if (t >= tilesM * tilesN) return;
  int tm = t / tilesN, tn = t % tilesN;
  int hi = lane >> 4;
  int rc = lane & 15;
  const _Float16* arow = A + (size_t)(tm * 16 + rc) * K + (hi ? 8 : 0);
  int gn = tn * 16 + rc;
  int gnc = gn < N ? gn : (N - 1);                 // clamp: always-valid loads
  const _Float16* brow = Bt + (size_t)gnc * K + (hi ? 8 : 0);
  v8f acc = {};
  for (int kb = 0; kb < K; kb += 32) {
    v16h af, bf;
    *(v8h*)&af       = *(const v8h*)(arow + kb);
    *((v8h*)&af + 1) = *(const v8h*)(arow + kb + 16);
    *(v8h*)&bf       = *(const v8h*)(brow + kb);
    *((v8h*)&bf + 1) = *(const v8h*)(brow + kb + 16);
    acc = __builtin_amdgcn_wmma_f32_16x16x32_f16(false, af, false, bf, (short)0, acc,
                                                 false, false);
  }
  int col = tn * 16 + rc;
  if (col < N) {
    float bv = bias ? bias[col] : 0.f;
    for (int vg = 0; vg < 8; ++vg) {
      int row = tm * 16 + vg + hi * 8;
      float val = acc[vg] + bv;
      if (act == 1) val = fmaxf(val, 0.f);
      else if (act == 2) val = 1.f / (1.f + __expf(-val));
      if (umode) {
        // row = b*96+l ; col = oc = g*8+d -> u[b, g*96+l, d]
        int b = row / LOUT, l = row % LOUT;
        int g = col >> 3, d = col & 7;
        Ch[(((size_t)b * NROUTES) + (size_t)g * LOUT + l) * 8 + d] = (_Float16)val;
      } else {
        if (Ch) Ch[(size_t)row * N + col] = (_Float16)val;
        if (Cf) Cf[(size_t)row * N + col] = val;
      }
    }
  }
}

// ---------------- routing "s" GEMM ----------------
// s[b,c,o] += sum_k coup(b,c,k/8) * u[b,k] * Wt[c,o,k] ; K = 24576, split 16 ways.
// coup==null -> uniform 1/50 applied at the end. Two inner loops so the uniform
// branch is hoisted; coupling applied as packed v8h multiplies (v_pk_mul_f16).
__global__ __launch_bounds__(128) void wmma_route_s_k(
    const _Float16* __restrict__ uh, const _Float16* __restrict__ Wt,
    const _Float16* __restrict__ coup, float* __restrict__ s) {
  const int KS = 16, STEPS = 48;  // 16*48*32 = 24576
  int lane = threadIdx.x & 31;
  int wv = threadIdx.x >> 5;
  int t = blockIdx.x * 4 + wv;     // 0..3199 = c(50) * mt(4) * ks(16)
  int c = t / (4 * KS);
  int rem = t % (4 * KS);
  int mt = rem / KS, ks = rem % KS;
  int hi = lane >> 4;
  int rc = lane & 15;
  int b = mt * 16 + rc;
  int kb0 = ks * STEPS * 32;
  const _Float16* arow = uh + (size_t)b * KROUTE + kb0 + (hi ? 8 : 0);
  const _Float16* brow = Wt + ((size_t)c * 16 + rc) * KROUTE + kb0 + (hi ? 8 : 0);
  v8f acc = {};
  if (coup) {
    // coup index for the two 8-half chunks: route r = (global k)>>3
    const _Float16* crow = coup + (size_t)c * NROUTES * BATCH + b;
    int r0 = (kb0 + (hi ? 8 : 0)) >> 3;   // route of chunk0 at step 0
    for (int st = 0; st < STEPS; ++st) {
      v16h af, bf;
      v8h a0 = *(const v8h*)(arow + st * 32);
      v8h a1 = *(const v8h*)(arow + st * 32 + 16);
      _Float16 s0 = crow[(size_t)(r0 + st * 4) * BATCH];
      _Float16 s1 = crow[(size_t)(r0 + st * 4 + 2) * BATCH];
      v8h s0v = {s0, s0, s0, s0, s0, s0, s0, s0};
      v8h s1v = {s1, s1, s1, s1, s1, s1, s1, s1};
      a0 = a0 * s0v;
      a1 = a1 * s1v;
      *(v8h*)&af       = a0;
      *((v8h*)&af + 1) = a1;
      *(v8h*)&bf       = *(const v8h*)(brow + st * 32);
      *((v8h*)&bf + 1) = *(const v8h*)(brow + st * 32 + 16);
      acc = __builtin_amdgcn_wmma_f32_16x16x32_f16(false, af, false, bf, (short)0, acc,
                                                   false, false);
    }
  } else {
    for (int st = 0; st < STEPS; ++st) {
      v16h af, bf;
      *(v8h*)&af       = *(const v8h*)(arow + st * 32);
      *((v8h*)&af + 1) = *(const v8h*)(arow + st * 32 + 16);
      *(v8h*)&bf       = *(const v8h*)(brow + st * 32);
      *((v8h*)&bf + 1) = *(const v8h*)(brow + st * 32 + 16);
      acc = __builtin_amdgcn_wmma_f32_16x16x32_f16(false, af, false, bf, (short)0, acc,
                                                   false, false);
    }
  }
  float sc = coup ? 1.f : (1.f / 50.f);
  for (int vg = 0; vg < 8; ++vg) {
    int bb = mt * 16 + vg + hi * 8;
    unsafeAtomicAdd(&s[((size_t)bb * NCAPS + c) * OUTD + rc], acc[vg] * sc);
  }
}

// v = squash(s): per (b,c) 16-vector
__global__ void squash_k(const float* __restrict__ s, float* __restrict__ v) {
  int idx = blockIdx.x * 256 + threadIdx.x;
  if (idx >= BATCH * NCAPS) return;
  const float* sp = s + (size_t)idx * OUTD;
  float n2 = 0.f;
  for (int o = 0; o < OUTD; ++o) n2 += sp[o] * sp[o];
  float n = sqrtf(n2);
  float sc = (n2 / (1.f + n2)) / (n + 1e-8f);
  for (int o = 0; o < OUTD; ++o) v[(size_t)idx * OUTD + o] = sc * sp[o];
}

// b_log[c,r,b] (+)= sum_{i,o} u[b,r,i] * W[c,r,i,o] * v[b,c,o]
__global__ __launch_bounds__(64) void blog_update_k(
    const _Float16* __restrict__ uh, const float* __restrict__ W,
    const float* __restrict__ v, float* __restrict__ blog, int accum) {
  __shared__ float Ws[128];
  int c = blockIdx.x / NROUTES;
  int r = blockIdx.x % NROUTES;
  int b = threadIdx.x;  // 0..63
  const float* Wp = W + ((size_t)c * NROUTES + r) * 128;
  Ws[b] = Wp[b];
  Ws[b + 64] = Wp[b + 64];
  __syncthreads();
  float uf[8];
  const _Float16* up = uh + ((size_t)b * NROUTES + r) * 8;
  for (int i = 0; i < 8; ++i) uf[i] = (float)up[i];
  float vl[16];
  const float* vp = v + ((size_t)b * NCAPS + c) * OUTD;
  for (int o = 0; o < OUTD; ++o) vl[o] = vp[o];
  float acc = 0.f;
  for (int i = 0; i < 8; ++i) {
    float tmp = 0.f;
    for (int o = 0; o < OUTD; ++o) tmp += Ws[i * 16 + o] * vl[o];
    acc += uf[i] * tmp;
  }
  size_t bi = ((size_t)c * NROUTES + r) * BATCH + b;
  blog[bi] = accum ? (blog[bi] + acc) : acc;
}

// softmax over c (axis=1) per (b,r); coup[c,r,b] f16
__global__ void softmax_c_k(const float* __restrict__ blog, _Float16* __restrict__ coup) {
  int idx = blockIdx.x * 256 + threadIdx.x;
  if (idx >= NROUTES * BATCH) return;
  int r = idx >> 6, b = idx & 63;
  float t[NCAPS];
  float mx = -3.0e38f;
  for (int c = 0; c < NCAPS; ++c) {
    t[c] = blog[((size_t)c * NROUTES + r) * BATCH + b];
    mx = fmaxf(mx, t[c]);
  }
  float sum = 0.f;
  for (int c = 0; c < NCAPS; ++c) { t[c] = __expf(t[c] - mx); sum += t[c]; }
  float inv = 1.f / sum;
  for (int c = 0; c < NCAPS; ++c)
    coup[((size_t)c * NROUTES + r) * BATCH + b] = (_Float16)(t[c] * inv);
}

// lengths -> d_out[0..3199], argmax mask, masked v -> A1(B,800) f16
__global__ void lengths_mask_k(const float* __restrict__ v, float* __restrict__ out,
                               _Float16* __restrict__ A1) {
  int b = blockIdx.x, tid = threadIdx.x;
  __shared__ float lens[NCAPS];
  __shared__ int amax;
  if (tid < NCAPS) {
    const float* vp = v + ((size_t)b * NCAPS + tid) * OUTD;
    float n2 = 0.f;
    for (int o = 0; o < OUTD; ++o) n2 += vp[o] * vp[o];
    float n = sqrtf(n2);
    lens[tid] = n;
    out[b * NCAPS + tid] = n;
  }
  __syncthreads();
  if (tid == 0) {
    float best = lens[0]; int bi = 0;
    for (int c = 1; c < NCAPS; ++c)
      if (lens[c] > best) { best = lens[c]; bi = c; }
    amax = bi;
  }
  __syncthreads();
  for (int j = tid; j < NCAPS * OUTD; j += 256) {
    int c = j >> 4, o = j & 15;
    float val = (c == amax) ? v[((size_t)b * NCAPS + c) * OUTD + o] : 0.f;
    A1[(size_t)b * 800 + j] = (_Float16)val;
  }
}

// ---------------- host orchestration ----------------
extern "C" void kernel_launch(void* const* d_in, const int* in_sizes, int n_in,
                              void* d_out, int out_size, void* d_ws, size_t ws_size,
                              hipStream_t stream) {
  const int*   x      = (const int*)d_in[0];
  const float* emb    = (const float*)d_in[1];
  const float* conv_w = (const float*)d_in[2];
  const float* conv_b = (const float*)d_in[3];
  const float* W      = (const float*)d_in[4];
  const float* w1     = (const float*)d_in[5];
  const float* b1     = (const float*)d_in[6];
  const float* w2     = (const float*)d_in[7];
  const float* b2     = (const float*)d_in[8];
  const float* w3     = (const float*)d_in[9];
  const float* b3     = (const float*)d_in[10];
  float* out = (float*)d_out;

  char* base = (char*)d_ws;
  size_t off = 0;
  auto take = [&](size_t bytes) -> void* {
    void* p = base + off;
    off = (off + bytes + 255) & ~(size_t)255;
    return p;
  };
  _Float16* uh   = (_Float16*)take((size_t)BATCH * NROUTES * 8 * 2);       // 3.1 MB
  _Float16* Ac   = (_Float16*)take((size_t)BATCH * LOUT * CONVK * 2);      // 7.1 MB
  _Float16* Bc   = (_Float16*)take((size_t)256 * CONVK * 2);
  _Float16* w1t  = (_Float16*)take((size_t)512 * 800 * 2);
  _Float16* w2t  = (_Float16*)take((size_t)1024 * 512 * 2);
  _Float16* w3t  = (_Float16*)take((size_t)RECON * 1024 * 2);              // 38.9 MB
  _Float16* Wt   = (_Float16*)take((size_t)NCAPS * 16 * KROUTE * 2);       // 39.3 MB
  float*    blog = (float*)take((size_t)NCAPS * NROUTES * BATCH * 4);      // 39.3 MB
  _Float16* coup = (_Float16*)take((size_t)NCAPS * NROUTES * BATCH * 2);   // 19.7 MB
  float*    sbuf = (float*)take((size_t)BATCH * NCAPS * OUTD * 4);
  float*    vbuf = (float*)take((size_t)BATCH * NCAPS * OUTD * 4);
  _Float16* A1   = (_Float16*)take((size_t)BATCH * 800 * 2);
  _Float16* A2   = (_Float16*)take((size_t)BATCH * 512 * 2);
  _Float16* A3   = (_Float16*)take((size_t)BATCH * 1024 * 2);

  dim3 tb(32, 8);
  // weight conversion / transposition (f32 -> f16, (K,N)->(N,K))
  cvt_f16_k<<<(256 * CONVK + 255) / 256, 256, 0, stream>>>(conv_w, Bc, 256 * CONVK);
  transpose_cvt_k<<<dim3(16, 25, 1), tb, 0, stream>>>(w1, w1t, 800, 512);
  transpose_cvt_k<<<dim3(32, 16, 1), tb, 0, stream>>>(w2, w2t, 512, 1024);
  transpose_cvt_k<<<dim3((RECON + 31) / 32, 32, 1), tb, 0, stream>>>(w3, w3t, 1024, RECON);
  transpose_cvt_k<<<dim3(1, KROUTE / 32, NCAPS), tb, 0, stream>>>(W, Wt, KROUTE, 16);

  // embedding gather + im2col, then conv GEMM -> u (f16, capsule layout), bias+relu fused
  build_patches_k<<<(BATCH * LOUT * CONVK + 255) / 256, 256, 0, stream>>>(x, emb, Ac);
  wmma_gemm_k<<<(384 * 16) / 4, 128, 0, stream>>>(Ac, Bc, conv_b, nullptr, uh,
                                                  BATCH * LOUT, 256, CONVK, 1, 1);

  // dynamic routing (3 iters): s-GEMM (WMMA, K=24576, 16-way K-split) + squash,
  // scalar b_log update + softmax between iterations
  auto routing_s = [&](const _Float16* cp) {
    zero_f32_k<<<(BATCH * NCAPS * OUTD + 255) / 256, 256, 0, stream>>>(sbuf,
                                                                       BATCH * NCAPS * OUTD);
    wmma_route_s_k<<<(NCAPS * 4 * 16) / 4, 128, 0, stream>>>(uh, Wt, cp, sbuf);
    squash_k<<<(BATCH * NCAPS + 255) / 256, 256, 0, stream>>>(sbuf, vbuf);
  };
  routing_s(nullptr);                                                   // iter 0
  blog_update_k<<<NCAPS * NROUTES, 64, 0, stream>>>(uh, W, vbuf, blog, 0);
  softmax_c_k<<<(NROUTES * BATCH + 255) / 256, 256, 0, stream>>>(blog, coup);
  routing_s(coup);                                                      // iter 1
  blog_update_k<<<NCAPS * NROUTES, 64, 0, stream>>>(uh, W, vbuf, blog, 1);
  softmax_c_k<<<(NROUTES * BATCH + 255) / 256, 256, 0, stream>>>(blog, coup);
  routing_s(coup);                                                      // iter 2 (final v)

  // capsule lengths -> out[0..3199]; masked v -> A1
  lengths_mask_k<<<BATCH, 256, 0, stream>>>(vbuf, out, A1);

  // reconstruction MLP (WMMA GEMMs, fused bias + relu/sigmoid)
  wmma_gemm_k<<<(4 * 32) / 4, 128, 0, stream>>>(A1, w1t, b1, nullptr, A2, BATCH, 512, 800, 1, 0);
  wmma_gemm_k<<<(4 * 64) / 4, 128, 0, stream>>>(A2, w2t, b2, nullptr, A3, BATCH, 1024, 512, 1, 0);
  wmma_gemm_k<<<(4 * 1188) / 4, 128, 0, stream>>>(A3, w3t, b3, out + BATCH * NCAPS, nullptr,
                                                  BATCH, RECON, 1024, 2, 0);
}
